// Network_49873160241834
// MI455X (gfx1250) — matrix operand order repaired and checked by reading, not compiled, not run on hardware
//
#include <hip/hip_runtime.h>
#include <hip/hip_bf16.h>

// Problem constants (from reference setup_inputs)
#define DT_   0.001f
#define B_    256
#define F_    10
#define T_    8192
#define TSEG  1024               // t-extent handled by one block
#define TC    512                // t-extent scanned by one thread
#define WARM  128                // warm-up steps (state decay 0.8^128 ~ 4e-13)
#define NSEG  (T_ / TSEG)        // 8 segments per batch row
#define NTILE (TSEG / 16)        // 64 WMMA tiles per segment
#define CPT   (TSEG / TC)        // 2 chunks per feature per block
#define SCANT (F_ * CPT)         // 20 scan threads per block
#define KPAD  16                 // K padded to 16 (cols 10..15 = finite zeros)

typedef _Float16 v16h __attribute__((ext_vector_type(16)));
typedef float    v8f  __attribute__((ext_vector_type(8)));

__global__ __launch_bounds__(64) void snn_fused_kernel(
    const float* __restrict__ x,    // (B, F, T)
    const float* __restrict__ tau,  // (3,)
    const float* __restrict__ vth,  // (3,)
    const float* __restrict__ cw,   // conv_w flattened (3,)
    const float* __restrict__ cb,   // conv_b (1,)
    const float* __restrict__ lw,   // lin_w (2,10)
    const float* __restrict__ lb,   // lin_b (2,)
    float* __restrict__ out)        // (B, 2, T)
{
    // Transposed fp16 staging tile: convsT[t_local][k], 1024*16*2 = 32 KB LDS.
    // Row stride 32 B -> one lane's B-fragment is 16 contiguous halves.
    __shared__ _Float16 convsT[TSEG][KPAD];

    const int b     = blockIdx.x / NSEG;
    const int seg   = blockIdx.x % NSEG;
    const int tseg0 = seg * TSEG;
    const int tid   = threadIdx.x;

    // Zero the K-pad columns (10..15): WMMA needs finite values there
    // (their weights in A are zero, so any finite value contributes 0).
    for (int t = tid; t < TSEG; t += 64) {
        *(unsigned int*)&convsT[t][10] = 0u;
        *(unsigned int*)&convsT[t][12] = 0u;
        *(unsigned int*)&convsT[t][14] = 0u;
    }

    // ---------------- Phase 1: chunk-parallel LIF scan -> LDS (fp16) -------
    if (tid < SCANT) {
        const int f  = tid / CPT;
        const int jc = tid % CPT;
        const int t0 = tseg0 + jc * TC;
        int tw = t0 - WARM; if (tw < 0) tw = 0;

        const float a0 = DT_ * tau[0], a1 = DT_ * tau[1], a2 = DT_ * tau[2];
        const float th0 = vth[0], th1 = vth[1], th2 = vth[2];
        const float w0 = cw[0], w1 = cw[1], w2 = cw[2];
        const float cbv = cb[0];

        const float* __restrict__ xrow = x + ((size_t)b * F_ + f) * T_;
        float v0 = 0.f, v1 = 0.f, v2 = 0.f;

        // Warm-up: run dynamics (incl. spike resets) without emitting output
        for (int t = tw; t < t0; t += 4) {
            float4 xv = *(const float4*)(xrow + t);
            float xe[4] = {xv.x, xv.y, xv.z, xv.w};
#pragma unroll
            for (int e = 0; e < 4; ++e) {
                v0 = fmaf(a0, xe[e] - v0, v0); v0 = (v0 > th0) ? 0.f : v0;
                v1 = fmaf(a1, xe[e] - v1, v1); v1 = (v1 > th1) ? 0.f : v1;
                v2 = fmaf(a2, xe[e] - v2, v2); v2 = (v2 > th2) ? 0.f : v2;
            }
        }
        // Main scan: emit conv = sum_c w_c*z_c + conv_b as fp16 into LDS
        for (int t = t0; t < t0 + TC; t += 4) {
            float4 xv = *(const float4*)(xrow + t);
            __builtin_prefetch(xrow + t + 64, 0, 0);  // global_prefetch_b8
            float xe[4] = {xv.x, xv.y, xv.z, xv.w};
            const int tl = t - tseg0;
#pragma unroll
            for (int e = 0; e < 4; ++e) {
                float acc = cbv;
                v0 = fmaf(a0, xe[e] - v0, v0);
                bool s0 = v0 > th0; acc += s0 ? w0 : 0.f; v0 = s0 ? 0.f : v0;
                v1 = fmaf(a1, xe[e] - v1, v1);
                bool s1 = v1 > th1; acc += s1 ? w1 : 0.f; v1 = s1 ? 0.f : v1;
                v2 = fmaf(a2, xe[e] - v2, v2);
                bool s2 = v2 > th2; acc += s2 ? w2 : 0.f; v2 = s2 ? 0.f : v2;
                convsT[tl + e][f] = (_Float16)acc;   // ds_store_b16
            }
        }
    }
    __syncthreads();

    // ---------------- Phase 2: 2x10 linear layer via v_wmma_f32_16x16x32_f16
    const int lane = tid & 31;
    const int wv   = tid >> 5;     // wave in block (0..1)
    const int m    = lane & 15;    // A row & B/D column (t within tile)
    const int kh   = lane >> 4;    // lane half selects K-range

    // A-matrix (16x32 fp16): rows m<2 = lin_w, rest zero. Contiguous K order:
    // lane<16: elems 0..15 -> K0..15; lane>=16: elems 0..15 -> K16..31.
    // All K>=10 weights are exactly zero -> B's K>=10 content is don't-care
    // (as long as finite).
    v16h afrag;
#pragma unroll
    for (int i = 0; i < 16; ++i) {
        int k = i + kh * 16;
        float wf = (m < 2 && k < F_) ? lw[m * F_ + k] : 0.f;
        afrag[i] = (_Float16)wf;
    }
    const float lb0 = lb[0], lb1 = lb[1];
    float* __restrict__ out0 = out + ((size_t)b * 2 + 0) * T_ + tseg0;
    float* __restrict__ out1 = out0 + T_;

#pragma unroll 2
    for (int tile = wv; tile < NTILE; tile += 2) {
        const int tloc = tile * 16 + m;
        // Unconditional 32-byte LDS load for ALL lanes (two ds_load_b128):
        // lanes 0-15 supply B rows K=0..15; lanes 16-31 nominally supply
        // K=16..31 but read the same finite data -> zero contribution via A.
        v16h bfrag = *(const v16h*)&convsT[tloc][0];
        v8f cacc = {0.f, 0.f, 0.f, 0.f, 0.f, 0.f, 0.f, 0.f};
        v8f d = __builtin_amdgcn_wmma_f32_16x16x32_f16(
            /*neg_a=*/false, afrag, /*neg_b=*/false, bfrag,
            /*c_mod=*/(short)0, cacc, /*reuse_a=*/false, /*reuse_b=*/false);
        // C/D layout: lane<16 VGPR r -> row M=r. Rows 0,1 are the outputs.
        if (kh == 0) {
            out0[tloc] = d[0] + lb0;
            out1[tloc] = d[1] + lb1;
        }
    }
}

extern "C" void kernel_launch(void* const* d_in, const int* in_sizes, int n_in,
                              void* d_out, int out_size, void* d_ws, size_t ws_size,
                              hipStream_t stream) {
    (void)in_sizes; (void)n_in; (void)d_ws; (void)ws_size; (void)out_size;
    const float* x   = (const float*)d_in[0];
    const float* tau = (const float*)d_in[1];
    const float* vth = (const float*)d_in[2];
    const float* cw  = (const float*)d_in[3];
    const float* cb  = (const float*)d_in[4];
    const float* lw  = (const float*)d_in[5];
    const float* lb  = (const float*)d_in[6];
    float* out = (float*)d_out;

    dim3 grid(B_ * NSEG);   // 2048 blocks: one (batch, 1024-step segment) each
    dim3 block(64);         // 2 waves: 20 scan lanes, then 2 full WMMA waves
    snn_fused_kernel<<<grid, block, 0, stream>>>(x, tau, vth, cw, cb, lw, lb, out);
}
